// EfficientNetB0_Mamba_BiGRU_SelfAttention_V2_91225105367713
// MI455X (gfx1250) — compile-verified, matrix operand order checked by reading, not run on hardware
//
#include <hip/hip_runtime.h>
#include <hip/hip_bf16.h>
#include <math.h>

typedef __bf16 bf16;
typedef __attribute__((ext_vector_type(8)))  __bf16 v8bf;
typedef __attribute__((ext_vector_type(16))) __bf16 v16bf;
typedef __attribute__((ext_vector_type(8)))  float  v8f;

// ---------------------------------------------------------------------------
// WMMA helpers (CDNA5 wave32, 16x16x32 bf16 -> f32)
// ---------------------------------------------------------------------------
__device__ __forceinline__ v8f wmma_bf16(v16bf a, v16bf b, v8f c) {
  return __builtin_amdgcn_wmma_f32_16x16x32_bf16(
      /*neg_a=*/false, a, /*neg_b=*/false, b,
      /*c_mod=*/(short)0, c, /*reuse_a=*/false, /*reuse_b=*/false);
}

// Load one 16x32 bf16 fragment (A- or B-operand) for this lane.
// rowp points at the start of the 32-element K-row for this lane's M (or N).
// ISA layout: lanes 0-15 hold K=[0..7]+[16..23]; lanes 16-31 hold K=[8..15]+[24..31].
__device__ __forceinline__ v16bf load_frag(const bf16* rowp, int lane) {
  const int khalf = (lane >> 4) & 1;
  v8bf lo = *(const v8bf*)(rowp + khalf * 8);
  v8bf hi = *(const v8bf*)(rowp + khalf * 8 + 16);
  return __builtin_shufflevector(lo, hi, 0, 1, 2, 3, 4, 5, 6, 7,
                                 8, 9, 10, 11, 12, 13, 14, 15);
}

// ---------------------------------------------------------------------------
// CDNA5 async global->LDS copy (ASYNCcnt-tracked, no VGPR staging).
// Generic->LDS address: low 32 bits of the generic pointer are the LDS offset.
// ---------------------------------------------------------------------------
__device__ __forceinline__ void async_copy_b128(bf16* lds_dst, const bf16* gsrc) {
  unsigned lds_off = (unsigned)(unsigned long long)lds_dst;
  asm volatile("global_load_async_to_lds_b128 %0, %1, off"
               :: "v"(lds_off), "v"(gsrc)
               : "memory");
}

__device__ __forceinline__ void wait_async0() {
#if __has_builtin(__builtin_amdgcn_s_wait_asynccnt)
  __builtin_amdgcn_s_wait_asynccnt(0);
#else
  asm volatile("s_wait_asynccnt 0x0" ::: "memory");
#endif
}

// ---------------------------------------------------------------------------
// Elementwise f32 -> bf16 conversion
// ---------------------------------------------------------------------------
__global__ void k_cvt(const float* __restrict__ in, bf16* __restrict__ out, long n) {
  long i = (long)blockIdx.x * 256 + threadIdx.x;
  if (i < n) out[i] = (bf16)in[i];
}

// ---------------------------------------------------------------------------
// Generic batched WMMA GEMM:
//   C[z, m, n] = alpha * sum_k A[z,m,k] * W(n,k)  (+ bias[n])
// W layout: kn==0 -> W is [N x K] row-major ("x @ W^T", PyTorch weight layout)
//           kn==1 -> W is [K x N] row-major ("P @ V")
// Workgroup tile 128x128, 8 waves (2x4), wave tile 64x32 (8 WMMA / K-step):
// this keeps the accumulator footprint at 64 VGPRs/wave so the hot loop stays
// spill-free (the 64x64 variant spilled accumulators to scratch every K-step).
// A/W tiles double-buffered in LDS and filled with global_load_async_to_lds,
// overlapping the fill of buffer k+1 with WMMA on buffer k.
// Writes f32 (Cf) and/or bf16 (Cb) output.
// ---------------------------------------------------------------------------
__global__ void k_gemm(const bf16* __restrict__ A, long lda, long sAz,
                       const bf16* __restrict__ W, long ldw, long sWz,
                       float* __restrict__ Cf, bf16* __restrict__ Cb,
                       long ldc, long sCz,
                       const float* __restrict__ bias, float alpha,
                       int K, int kn) {
  __shared__ __align__(16) bf16 As[2][128][32];
  __shared__ __align__(16) bf16 Bs[2][128][32];  // N-major, K-contiguous
  const int tid  = threadIdx.x;
  const int lane = tid & 31;
  const int w    = tid >> 5;
  const int wm   = w >> 2;   // 0..1  -> 64 rows
  const int wn   = w & 3;    // 0..3  -> 32 cols
  const long m0  = (long)blockIdx.y * 128;
  const long n0  = (long)blockIdx.x * 128;
  const long z   = blockIdx.z;
  const bf16* Ab = A + z * sAz;
  const bf16* Wb = W + z * sWz;

  // staging thread mappings (128x32 tile: 2 x 16B per thread)
  const int sr = tid >> 1, sc = (tid & 1) * 16;
  const int kk = tid >> 3, nb = (tid & 7) * 16;   // W(KN): 32 k x 8 n-blocks

  v8f acc[4][2];
  for (int i = 0; i < 4; ++i)
    for (int j = 0; j < 2; ++j)
      for (int v = 0; v < 8; ++v) acc[i][j][v] = 0.f;

  // ---- stage k0 = 0 into buffer 0 ----
  {
    const bf16* srcA = Ab + (m0 + sr) * lda + sc;
    async_copy_b128(&As[0][sr][sc], srcA);
    async_copy_b128(&As[0][sr][sc + 8], srcA + 8);
    if (!kn) {
      const bf16* srcW = Wb + (n0 + sr) * ldw + sc;
      async_copy_b128(&Bs[0][sr][sc], srcW);
      async_copy_b128(&Bs[0][sr][sc + 8], srcW + 8);
    } else {
      const bf16* srcW = Wb + (long)kk * ldw + n0 + nb;
      v8bf t0 = *(const v8bf*)(srcW);
      v8bf t1 = *(const v8bf*)(srcW + 8);
#pragma unroll
      for (int e = 0; e < 8; ++e) {
        Bs[0][nb + e][kk]     = t0[e];
        Bs[0][nb + 8 + e][kk] = t1[e];
      }
    }
  }
  wait_async0();
  __syncthreads();

  int cur = 0;
  for (int k0 = 0; k0 < K; k0 += 32) {
    const int nxt = cur ^ 1;
    // ---- fill next K-slab into the other buffer (async, overlapped) ----
    if (k0 + 32 < K) {
      const int kn0 = k0 + 32;
      const bf16* srcA = Ab + (m0 + sr) * lda + kn0 + sc;
      async_copy_b128(&As[nxt][sr][sc], srcA);
      async_copy_b128(&As[nxt][sr][sc + 8], srcA + 8);
      if (!kn) {
        const bf16* srcW = Wb + (n0 + sr) * ldw + kn0 + sc;
        async_copy_b128(&Bs[nxt][sr][sc], srcW);
        async_copy_b128(&Bs[nxt][sr][sc + 8], srcW + 8);
      } else {
        const bf16* srcW = Wb + (long)(kn0 + kk) * ldw + n0 + nb;
        v8bf t0 = *(const v8bf*)(srcW);
        v8bf t1 = *(const v8bf*)(srcW + 8);
#pragma unroll
        for (int e = 0; e < 8; ++e) {
          Bs[nxt][nb + e][kk]     = t0[e];
          Bs[nxt][nb + 8 + e][kk] = t1[e];
        }
      }
    }

    // ---- compute on current buffer: 8 WMMA per wave per K-step ----
    v16bf afr[4];
#pragma unroll
    for (int i = 0; i < 4; ++i)
      afr[i] = load_frag(&As[cur][wm * 64 + i * 16 + (lane & 15)][0], lane);
#pragma unroll
    for (int j = 0; j < 2; ++j) {
      v16bf bfr = load_frag(&Bs[cur][wn * 32 + j * 16 + (lane & 15)][0], lane);
#pragma unroll
      for (int i = 0; i < 4; ++i) acc[i][j] = wmma_bf16(afr[i], bfr, acc[i][j]);
    }

    wait_async0();       // our async fills of `nxt` have landed in LDS
    __syncthreads();     // everyone done: reads of `cur`, fills of `nxt`
    cur = nxt;
  }

  // ---- epilogue: C/D 16x16 layout: row m = v + 8*(lane>=16), col n = lane&15
  for (int i = 0; i < 4; ++i)
    for (int j = 0; j < 2; ++j) {
      const long mbase = m0 + wm * 64 + i * 16 + ((lane >> 4) << 3);
      const long n     = n0 + wn * 32 + j * 16 + (lane & 15);
      const float bv   = bias ? bias[n] : 0.f;
#pragma unroll
      for (int v = 0; v < 8; ++v) {
        const float val = acc[i][j][v] * alpha + bv;
        const long idx  = z * sCz + (mbase + v) * ldc + n;
        if (Cf) Cf[idx] = val;
        if (Cb) Cb[idx] = (bf16)val;
      }
    }
}

// ---------------------------------------------------------------------------
// Mamba scan: h_t = gelu_erf(h_{t-1} @ A^T + u_t), 16 batch rows per WG.
// State held in LDS (bf16, feeds WMMA A-operand). A streamed from L2 (bf16 NK).
// Writes h_t as bf16 into hs[(b,s,h)] for the following y = hs @ C^T GEMM.
// ---------------------------------------------------------------------------
__global__ void k_mamba(const float* __restrict__ u, const bf16* __restrict__ Abf,
                        bf16* __restrict__ hs) {
  __shared__ __align__(16) bf16 hbf[16][512];
  const int tid  = threadIdx.x;
  const int lane = tid & 31;
  const int w    = tid >> 5;
  const int bb0  = blockIdx.x * 16;
  const int n0w  = w * 64;                 // 4 n-tiles per wave

  for (int i = tid; i < 16 * 512; i += 256) ((bf16*)hbf)[i] = (bf16)0.f;
  __syncthreads();

  for (int t = 0; t < 1024; ++t) {
    v8f acc[4];
    for (int j = 0; j < 4; ++j)
      for (int v = 0; v < 8; ++v) acc[j][v] = 0.f;

    for (int k0 = 0; k0 < 512; k0 += 32) {
      v16bf a = load_frag(&hbf[lane & 15][k0], lane);
#pragma unroll
      for (int j = 0; j < 4; ++j) {
        const int n = n0w + j * 16 + (lane & 15);
        v16bf b = load_frag(Abf + (long)n * 512 + k0, lane);
        acc[j] = wmma_bf16(a, b, acc[j]);
      }
    }
    __syncthreads();  // everyone done reading h_{t-1}

    const int mb = (lane >> 4) << 3;
#pragma unroll
    for (int j = 0; j < 4; ++j) {
      const int n = n0w + j * 16 + (lane & 15);
#pragma unroll
      for (int v = 0; v < 8; ++v) {
        const int m    = mb + v;
        const long idx = (((long)(bb0 + m)) * 1024 + t) * 512 + n;
        const float x  = acc[j][v] + u[idx];
        const float g  = 0.5f * x * (1.f + erff(x * 0.70710678118654752f));
        hbf[m][n] = (bf16)g;
        hs[idx]   = (bf16)g;
      }
    }
    __syncthreads();
  }
}

// ---------------------------------------------------------------------------
// GRU scan (one direction). 16 batch rows per WG. Input gates xi precomputed.
// Per step: gh = h @ w_hh^T (WMMA, M=16,N=768,K=256), then gate elementwise.
// Writes h into hcat[(b,s, dir_off+g)] as bf16 (input of Q/K/V GEMMs).
// ---------------------------------------------------------------------------
__global__ void k_gru(const float* __restrict__ xi, const bf16* __restrict__ whh,
                      const float* __restrict__ bhh, bf16* __restrict__ hcat,
                      int reverse, int dir_off) {
  __shared__ __align__(16) bf16 hbf[16][256];
  __shared__ float hf[16][256];
  __shared__ float gh[16][768];
  const int tid  = threadIdx.x;
  const int lane = tid & 31;
  const int w    = tid >> 5;
  const int bb0  = blockIdx.x * 16;
  const int n0w  = w * 96;                 // 6 n-tiles per wave

  for (int i = tid; i < 16 * 256; i += 256) {
    ((bf16*)hbf)[i] = (bf16)0.f;
    ((float*)hf)[i] = 0.f;
  }
  __syncthreads();

  for (int it = 0; it < 1024; ++it) {
    const int t = reverse ? (1023 - it) : it;

    v8f acc[6];
    for (int j = 0; j < 6; ++j)
      for (int v = 0; v < 8; ++v) acc[j][v] = 0.f;

    for (int k0 = 0; k0 < 256; k0 += 32) {
      v16bf a = load_frag(&hbf[lane & 15][k0], lane);
#pragma unroll
      for (int j = 0; j < 6; ++j) {
        const int n = n0w + j * 16 + (lane & 15);
        v16bf b = load_frag(whh + (long)n * 256 + k0, lane);
        acc[j] = wmma_bf16(a, b, acc[j]);
      }
    }
    const int mb = (lane >> 4) << 3;
#pragma unroll
    for (int j = 0; j < 6; ++j) {
      const int n  = n0w + j * 16 + (lane & 15);
      const float b = bhh[n];
#pragma unroll
      for (int v = 0; v < 8; ++v) gh[mb + v][n] = acc[j][v] + b;
    }
    __syncthreads();  // gh ready; all reads of hbf done

    for (int e = tid; e < 16 * 256; e += 256) {
      const int m = e >> 8, g = e & 255;
      const long base = (((long)(bb0 + m)) * 1024 + t) * 768;
      const float xr = xi[base + g];
      const float xz = xi[base + 256 + g];
      const float xn = xi[base + 512 + g];
      const float r  = 1.f / (1.f + expf(-(xr + gh[m][g])));
      const float z  = 1.f / (1.f + expf(-(xz + gh[m][256 + g])));
      const float nn = tanhf(xn + r * gh[m][512 + g]);
      const float h  = (1.f - z) * nn + z * hf[m][g];
      hf[m][g]  = h;
      hbf[m][g] = (bf16)h;
      hcat[(((long)(bb0 + m)) * 1024 + t) * 512 + dir_off + g] = (bf16)h;
    }
    __syncthreads();
  }
}

// ---------------------------------------------------------------------------
// LayerNorm over H=512, one row per block (256 threads, 2 elems/thread).
// Writes bf16 (feeds GRU input-gate GEMMs).
// ---------------------------------------------------------------------------
__global__ void k_ln(const float* __restrict__ y, const float* __restrict__ g,
                     const float* __restrict__ b, bf16* __restrict__ out) {
  __shared__ float red[256];
  const long row = blockIdx.x;
  const float* p = y + row * 512;
  const int t = threadIdx.x;
  const float v0 = p[t], v1 = p[t + 256];

  red[t] = v0 + v1;
  __syncthreads();
  for (int off = 128; off > 0; off >>= 1) {
    if (t < off) red[t] += red[t + off];
    __syncthreads();
  }
  const float mu = red[0] * (1.f / 512.f);
  __syncthreads();

  const float d0 = v0 - mu, d1 = v1 - mu;
  red[t] = d0 * d0 + d1 * d1;
  __syncthreads();
  for (int off = 128; off > 0; off >>= 1) {
    if (t < off) red[t] += red[t + off];
    __syncthreads();
  }
  const float rs = rsqrtf(red[0] * (1.f / 512.f) + 1e-5f);
  out[row * 512 + t]       = (bf16)(d0 * rs * g[t] + b[t]);
  out[row * 512 + t + 256] = (bf16)(d1 * rs * g[t + 256] + b[t + 256]);
}

// ---------------------------------------------------------------------------
// Row softmax over 1024 (scale pre-applied in scores GEMM). Writes bf16 P.
// ---------------------------------------------------------------------------
__global__ void k_softmax(const float* __restrict__ S, bf16* __restrict__ P) {
  __shared__ float red[256];
  const long row = blockIdx.x;
  const float* p = S + row * 1024;
  const int t = threadIdx.x;
  float v[4], mx = -3.402823466e38f;
#pragma unroll
  for (int i = 0; i < 4; ++i) {
    v[i] = p[t + 256 * i];
    mx = fmaxf(mx, v[i]);
  }
  red[t] = mx;
  __syncthreads();
  for (int off = 128; off > 0; off >>= 1) {
    if (t < off) red[t] = fmaxf(red[t], red[t + off]);
    __syncthreads();
  }
  mx = red[0];
  __syncthreads();
  float s = 0.f;
#pragma unroll
  for (int i = 0; i < 4; ++i) {
    v[i] = expf(v[i] - mx);
    s += v[i];
  }
  red[t] = s;
  __syncthreads();
  for (int off = 128; off > 0; off >>= 1) {
    if (t < off) red[t] += red[t + off];
    __syncthreads();
  }
  const float inv = 1.f / red[0];
#pragma unroll
  for (int i = 0; i < 4; ++i) P[row * 1024 + t + 256 * i] = (bf16)(v[i] * inv);
}

// ---------------------------------------------------------------------------
// Final head: out[row, o] = att[row,:512] . w_fc[o,:512] + b_fc[o], o in {0,1}.
// One wave per row (8 rows / block).
// ---------------------------------------------------------------------------
__global__ void k_proj(const float* __restrict__ att, const float* __restrict__ wfc,
                       const float* __restrict__ bfc, float* __restrict__ out) {
  const int lane = threadIdx.x & 31;
  const int w    = threadIdx.x >> 5;
  const long row = (long)blockIdx.x * 8 + w;
  const float* p = att + row * 512;
  float s0 = 0.f, s1 = 0.f;
  for (int i = lane; i < 512; i += 32) {
    const float x = p[i];
    s0 += x * wfc[i];
    s1 += x * wfc[512 + i];
  }
#pragma unroll
  for (int off = 16; off > 0; off >>= 1) {
    s0 += __shfl_xor(s0, off);
    s1 += __shfl_xor(s1, off);
  }
  if (lane == 0) {
    out[row * 2 + 0] = s0 + bfc[0];
    out[row * 2 + 1] = s1 + bfc[1];
  }
}

// ---------------------------------------------------------------------------
// Host-side orchestration
// ---------------------------------------------------------------------------
static inline void gemm(const bf16* A, long lda, long sAz,
                        const bf16* W, long ldw, long sWz,
                        float* Cf, bf16* Cb, long ldc, long sCz,
                        const float* bias, float alpha,
                        int M, int N, int K, int kn, int nz, hipStream_t s) {
  dim3 g(N / 128, M / 128, nz), b(256);
  k_gemm<<<g, b, 0, s>>>(A, lda, sAz, W, ldw, sWz, Cf, Cb, ldc, sCz, bias, alpha, K, kn);
}

extern "C" void kernel_launch(void* const* d_in, const int* in_sizes, int n_in,
                              void* d_out, int out_size, void* d_ws, size_t ws_size,
                              hipStream_t stream) {
  (void)in_sizes; (void)n_in; (void)out_size;
  const float* x     = (const float*)d_in[0];
  const float* Amat  = (const float*)d_in[1];
  const float* Bm    = (const float*)d_in[2];
  const float* Cm    = (const float*)d_in[3];
  const float* ln_g  = (const float*)d_in[4];
  const float* ln_b  = (const float*)d_in[5];
  const float* wihf  = (const float*)d_in[6];
  const float* whhf  = (const float*)d_in[7];
  const float* bihf  = (const float*)d_in[8];
  const float* bhhf  = (const float*)d_in[9];
  const float* wihb  = (const float*)d_in[10];
  const float* whhb  = (const float*)d_in[11];
  const float* bihb  = (const float*)d_in[12];
  const float* bhhb  = (const float*)d_in[13];
  const float* wq    = (const float*)d_in[14];
  const float* bq    = (const float*)d_in[15];
  const float* wk    = (const float*)d_in[16];
  const float* bk    = (const float*)d_in[17];
  const float* wv    = (const float*)d_in[18];
  const float* bv    = (const float*)d_in[19];
  const float* wfc   = (const float*)d_in[20];
  const float* bfc   = (const float*)d_in[21];

  char* ws = (char*)d_ws;
  constexpr size_t MB = 1ull << 20;
  // bf16 weight copies
  constexpr size_t O_BM   = 0;
  constexpr size_t O_A    = O_BM   + 524288;
  constexpr size_t O_C    = O_A    + 524288;
  constexpr size_t O_WIHF = O_C    + 524288;
  constexpr size_t O_WHHF = O_WIHF + 786432;
  constexpr size_t O_WIHB = O_WHHF + 393216;
  constexpr size_t O_WHHB = O_WIHB + 786432;
  constexpr size_t O_WQ   = O_WHHB + 393216;
  constexpr size_t O_WK   = O_WQ   + 524288;
  constexpr size_t O_WV   = O_WK   + 524288;
  // activations (regions reused once their producer/consumer chain is done)
  constexpr size_t O_XBF  = 8 * MB;      // x bf16 (64MB)
  constexpr size_t O_U    = 80 * MB;     // u f32 (128MB); later P bf16 (128MB)
  constexpr size_t O_HSBF = 224 * MB;    // hs bf16 (64MB)
  constexpr size_t O_YPRE = 304 * MB;    // y-pre-LN f32 (128MB); later att-out f32
  constexpr size_t O_YBF  = 456 * MB;    // y bf16 (64MB)
  constexpr size_t O_XIF  = 552 * MB;    // xi fwd f32 (192MB); later scores f32 (256MB)
  constexpr size_t O_XIB  = 776 * MB;    // xi bwd f32 (192MB)
  constexpr size_t O_HCAT = 1016 * MB;   // h concat bf16 (64MB)
  constexpr size_t O_QBF  = 1088 * MB;
  constexpr size_t O_KBF  = 1160 * MB;
  constexpr size_t O_VBF  = 1232 * MB;   // end ~1296MB
  if (ws_size < 1300 * MB) return;       // workspace too small; nothing sane to do

  const long NE_X = 64L * 1024 * 512;    // 33,554,432

#define CVT(src, dstoff, n) \
  k_cvt<<<(unsigned)(((n) + 255) / 256), 256, 0, stream>>>((src), (bf16*)(ws + (dstoff)), (long)(n))

  // 1) bf16 conversions
  CVT(x,    O_XBF,  NE_X);
  CVT(Amat, O_A,    512 * 512);
  CVT(Bm,   O_BM,   512 * 512);
  CVT(Cm,   O_C,    512 * 512);
  CVT(wihf, O_WIHF, 768 * 512);
  CVT(whhf, O_WHHF, 768 * 256);
  CVT(wihb, O_WIHB, 768 * 512);
  CVT(whhb, O_WHHB, 768 * 256);
  CVT(wq,   O_WQ,   512 * 512);
  CVT(wk,   O_WK,   512 * 512);
  CVT(wv,   O_WV,   512 * 512);
#undef CVT

  // 2) u = x @ Bm^T                (65536 x 512, K=512)
  gemm((bf16*)(ws + O_XBF), 512, 0, (bf16*)(ws + O_BM), 512, 0,
       (float*)(ws + O_U), nullptr, 512, 0, nullptr, 1.f,
       65536, 512, 512, 0, 1, stream);

  // 3) Mamba scan -> hs (bf16)
  k_mamba<<<4, 256, 0, stream>>>((const float*)(ws + O_U), (const bf16*)(ws + O_A),
                                 (bf16*)(ws + O_HSBF));

  // 4) y_pre = hs @ C^T
  gemm((bf16*)(ws + O_HSBF), 512, 0, (bf16*)(ws + O_C), 512, 0,
       (float*)(ws + O_YPRE), nullptr, 512, 0, nullptr, 1.f,
       65536, 512, 512, 0, 1, stream);

  // 5) LayerNorm -> y (bf16)
  k_ln<<<65536, 256, 0, stream>>>((const float*)(ws + O_YPRE), ln_g, ln_b,
                                  (bf16*)(ws + O_YBF));

  // 6/7) GRU input gates: xi = y @ w_ih^T + b_ih   (65536 x 768, K=512)
  gemm((bf16*)(ws + O_YBF), 512, 0, (bf16*)(ws + O_WIHF), 512, 0,
       (float*)(ws + O_XIF), nullptr, 768, 0, bihf, 1.f,
       65536, 768, 512, 0, 1, stream);
  gemm((bf16*)(ws + O_YBF), 512, 0, (bf16*)(ws + O_WIHB), 512, 0,
       (float*)(ws + O_XIB), nullptr, 768, 0, bihb, 1.f,
       65536, 768, 512, 0, 1, stream);

  // 8/9) GRU scans -> hcat (bf16, concat along feature dim)
  k_gru<<<4, 256, 0, stream>>>((const float*)(ws + O_XIF), (const bf16*)(ws + O_WHHF),
                               bhhf, (bf16*)(ws + O_HCAT), 0, 0);
  k_gru<<<4, 256, 0, stream>>>((const float*)(ws + O_XIB), (const bf16*)(ws + O_WHHB),
                               bhhb, (bf16*)(ws + O_HCAT), 1, 256);

  // 10/11) Q/K/V = hcat @ w^T + b   (bf16 outputs for attention)
  gemm((bf16*)(ws + O_HCAT), 512, 0, (bf16*)(ws + O_WQ), 512, 0,
       nullptr, (bf16*)(ws + O_QBF), 512, 0, bq, 1.f, 65536, 512, 512, 0, 1, stream);
  gemm((bf16*)(ws + O_HCAT), 512, 0, (bf16*)(ws + O_WK), 512, 0,
       nullptr, (bf16*)(ws + O_KBF), 512, 0, bk, 1.f, 65536, 512, 512, 0, 1, stream);
  gemm((bf16*)(ws + O_HCAT), 512, 0, (bf16*)(ws + O_WV), 512, 0,
       nullptr, (bf16*)(ws + O_VBF), 512, 0, bv, 1.f, 65536, 512, 512, 0, 1, stream);

  // 12) scores = (Q @ K^T) / sqrt(512)  per batch  (1024x1024, K=512, z=64)
  gemm((bf16*)(ws + O_QBF), 512, 1024L * 512, (bf16*)(ws + O_KBF), 512, 1024L * 512,
       (float*)(ws + O_XIF), nullptr, 1024, 1024L * 1024,
       nullptr, 0.04419417382415922f, 1024, 1024, 512, 0, 64, stream);

  // 13) softmax -> P (bf16)
  k_softmax<<<65536, 256, 0, stream>>>((const float*)(ws + O_XIF), (bf16*)(ws + O_U));

  // 14) att_out = P @ V  per batch  (1024x512, K=1024, V is [K x N])
  gemm((bf16*)(ws + O_U), 1024, 1024L * 1024, (bf16*)(ws + O_VBF), 512, 1024L * 512,
       (float*)(ws + O_YPRE), nullptr, 512, 1024L * 512,
       nullptr, 1.f, 1024, 512, 1024, 1, 64, stream);

  // 15) final head -> d_out (f32, (B,S,2))
  k_proj<<<8192, 256, 0, stream>>>((const float*)(ws + O_YPRE), wfc, bfc, (float*)d_out);
}
